// AnlaManifoldInpainter_v5_69561290326314
// MI455X (gfx1250) — compile-verified, matrix operand order checked by reading, not run on hardware
//
#include <hip/hip_runtime.h>
#include <hip/hip_bf16.h>

// ---------------------------------------------------------------------------
// Complex transformer forward for MI455X (gfx1250, wave32).
// GEMMs use V_WMMA_F32_16X16X32_F16 (f16 in / f32 accumulate).
// Complex GEMM uses the Gauss 3-mult decomposition:
//   p1=Ar*Br, p2=Ai*Bi, p3=(Ar+Ai)*(Br+Bi);  Cr=p1-p2, Ci=p3-p1-p2
// -> 6 WMMAs per 64x32 k-step instead of 8 (25% fewer matrix ops), and no
// negate->WMMA dependency chain (which cost hazard NOPs in the prev build).
// K is pipelined: next chunk global->regs while WMMAs run on the current LDS
// buffer, then regs->LDS (f32->f16 convert) into the alternate buffer
// (double-buffered, 72KB LDS, one barrier per chunk).
// ---------------------------------------------------------------------------

#define BB   32
#define SS   128
#define DD   1024
#define HH   16
#define DHD  64
#define FFD  4096
#define NBK  3
#define MTOK (BB * SS)       // 4096 tokens
#define LK   64              // K-chunk per LDS stage
#define LKP  72              // padded LDS row stride (halves) -> 144B rows

typedef _Float16 v16h __attribute__((ext_vector_type(16)));
typedef _Float16 h8v  __attribute__((ext_vector_type(8)));
typedef float    v8f  __attribute__((ext_vector_type(8)));

#define WMMA_F16(A_, B_, C_) \
  __builtin_amdgcn_wmma_f32_16x16x32_f16(false, (A_), false, (B_), (short)0, (C_), false, false)

// Load a v16h A/B fragment from an LDS row laid out [row][k] (stride LKP).
// CDNA5 16-bit operand layout: halves 0..7 = K[k4 + lhalf*8 .. +7],
//                              halves 8..15 = K[k4 + 16 + lhalf*8 .. +7].
__device__ __forceinline__ v16h ldfrag(const _Float16* row, int k4, int lhalf) {
  h8v lo = *(const h8v*)(row + k4 + lhalf * 8);
  h8v hi = *(const h8v*)(row + k4 + 16 + lhalf * 8);
  return __builtin_shufflevector(lo, hi, 0, 1, 2, 3, 4, 5, 6, 7,
                                         8, 9, 10, 11, 12, 13, 14, 15);
}

// ---------------------------------------------------------------------------
// Embedding gather + complex rotary: z[t,d] = emb[x[t],d] * e^{i*pos*invfreq_d}
// ---------------------------------------------------------------------------
__global__ __launch_bounds__(256) void k_embed_rot(const int* __restrict__ x,
                                                   const float2* __restrict__ emb,
                                                   float2* __restrict__ z) {
  int t = blockIdx.x;
  int s = t & (SS - 1);
  int idx = x[t];
  size_t eb = (size_t)idx * DD;
  size_t zb = (size_t)t * DD;
#pragma unroll
  for (int e = 0; e < 4; ++e) {
    int d = threadIdx.x + e * 256;
    float invf = __expf(-(float)d * (9.210340371976184f / (float)DD)); // ln(1e4)/D
    float sn, cs;
    __sincosf((float)s * invf, &sn, &cs);
    float2 v = emb[eb + d];
    z[zb + d] = make_float2(v.x * cs - v.y * sn, v.x * sn + v.y * cs);
  }
}

// ---------------------------------------------------------------------------
// Complex RMSNorm over D, real per-dim gain. One block per token.
// ---------------------------------------------------------------------------
__global__ __launch_bounds__(256) void k_crms(const float2* __restrict__ in,
                                              const float* __restrict__ g,
                                              float2* __restrict__ out) {
  size_t base = (size_t)blockIdx.x * DD;
  int tid = threadIdx.x;
  float2 v[4];
  float acc = 0.f;
#pragma unroll
  for (int e = 0; e < 4; ++e) {
    v[e] = in[base + tid + e * 256];
    acc += v[e].x * v[e].x + v[e].y * v[e].y;
  }
#pragma unroll
  for (int o = 16; o > 0; o >>= 1) acc += __shfl_xor(acc, o, 32);
  __shared__ float sred[8];
  if ((tid & 31) == 0) sred[tid >> 5] = acc;
  __syncthreads();
  float tot = 0.f;
#pragma unroll
  for (int w = 0; w < 8; ++w) tot += sred[w];
  float inv = rsqrtf(tot * (1.0f / (float)DD) + 1e-6f);
#pragma unroll
  for (int e = 0; e < 4; ++e) {
    int d = tid + e * 256;
    float gg = g[d] * inv;
    out[base + d] = make_float2(v[e].x * gg, v[e].y * gg);
  }
}

// ---------------------------------------------------------------------------
// PhaseTwist: h *= e^{i*|h|}
// ---------------------------------------------------------------------------
__global__ __launch_bounds__(256) void k_phase(float2* __restrict__ p, size_t n) {
  size_t i = (size_t)blockIdx.x * 256 + threadIdx.x;
  if (i >= n) return;
  float2 v = p[i];
  float mag = sqrtf(v.x * v.x + v.y * v.y);
  float s, c;
  __sincosf(mag, &s, &c);
  p[i] = make_float2(v.x * c - v.y * s, v.x * s + v.y * c);
}

// ---------------------------------------------------------------------------
// Generic complex GEMM: C[m,n] = sum_k A[m,k] * W[n,k]  (einsum 'md,nd->mn')
// flags bit0: C += resid ; flags bit1: C += bias[n]
// Grid: (N/64, M/64), 256 threads. Gauss 3-mult + double-buffered K pipeline.
// ---------------------------------------------------------------------------
__global__ __launch_bounds__(256) void k_cgemm(const float2* __restrict__ A,
                                               const float2* __restrict__ W,
                                               const float2* __restrict__ resid,
                                               const float2* __restrict__ bias,
                                               float2* __restrict__ C,
                                               int Kdim, int lda, int ldw, int ldc,
                                               int flags) {
  __shared__ __attribute__((aligned(16))) _Float16 sAr[2][64][LKP], sAi[2][64][LKP];
  __shared__ __attribute__((aligned(16))) _Float16 sBr[2][64][LKP], sBi[2][64][LKP];

  int n0 = blockIdx.x * 64, m0 = blockIdx.y * 64;
  int tid = threadIdx.x, lane = tid & 31, wave = tid >> 5;
  int wm = wave & 3, wn = wave >> 2;
  int lmod = lane & 15, lhalf = lane >> 4;
  int srow = tid >> 2, skg = (tid & 3) * 16;

  // Gauss accumulators: per n-tile {p1=ArBr, p2=AiBi, p3=(Ar+Ai)(Br+Bi)}
  v8f p1a = {0, 0, 0, 0, 0, 0, 0, 0}, p2a = {0, 0, 0, 0, 0, 0, 0, 0},
      p3a = {0, 0, 0, 0, 0, 0, 0, 0};
  v8f p1b = {0, 0, 0, 0, 0, 0, 0, 0}, p2b = {0, 0, 0, 0, 0, 0, 0, 0},
      p3b = {0, 0, 0, 0, 0, 0, 0, 0};

  const float4* ga4 = (const float4*)(A + (size_t)(m0 + srow) * lda + skg);
  const float4* gw4 = (const float4*)(W + (size_t)(n0 + srow) * ldw + skg);
  const int g4stride = LK / 2;      // LK float2 = LK/2 float4 per chunk step

  float4 ra[8], rw[8];
  // ---- prologue: stage chunk 0 into buffer 0 ----
#pragma unroll
  for (int j = 0; j < 8; ++j) { ra[j] = ga4[j]; rw[j] = gw4[j]; }
#pragma unroll
  for (int j = 0; j < 8; ++j) {
    sAr[0][srow][skg + 2 * j]     = (_Float16)ra[j].x;
    sAi[0][srow][skg + 2 * j]     = (_Float16)ra[j].y;
    sAr[0][srow][skg + 2 * j + 1] = (_Float16)ra[j].z;
    sAi[0][srow][skg + 2 * j + 1] = (_Float16)ra[j].w;
    sBr[0][srow][skg + 2 * j]     = (_Float16)rw[j].x;
    sBi[0][srow][skg + 2 * j]     = (_Float16)rw[j].y;
    sBr[0][srow][skg + 2 * j + 1] = (_Float16)rw[j].z;
    sBi[0][srow][skg + 2 * j + 1] = (_Float16)rw[j].w;
  }
  __syncthreads();

  int p = 0;
  for (int kc = 0; kc < Kdim; kc += LK) {
    const bool hasNext = (kc + LK) < Kdim;
    // ---- issue next chunk's global loads early (overlap with WMMAs) ----
    if (hasNext) {
      const float4* na = ga4 + (size_t)(kc / LK + 1) * g4stride;
      const float4* nw = gw4 + (size_t)(kc / LK + 1) * g4stride;
      if (kc + 2 * LK < Kdim) {                    // warm L2 two chunks ahead
        __builtin_prefetch(na + g4stride, 0, 1);   // global_prefetch_b8
        __builtin_prefetch(nw + g4stride, 0, 1);
      }
#pragma unroll
      for (int j = 0; j < 8; ++j) { ra[j] = na[j]; rw[j] = nw[j]; }
    }
    // ---- compute on current buffer ----
    const _Float16(*cAr)[LKP] = sAr[p];
    const _Float16(*cAi)[LKP] = sAi[p];
    const _Float16(*cBr)[LKP] = sBr[p];
    const _Float16(*cBi)[LKP] = sBi[p];
#pragma unroll
    for (int ks = 0; ks < 2; ++ks) {
      int k4 = ks * 32;
      int arow = wm * 16 + lmod;
      v16h ar = ldfrag(&cAr[arow][0], k4, lhalf);
      v16h ai = ldfrag(&cAi[arow][0], k4, lhalf);
      v16h as = ar + ai;                       // v_pk_add_f16
      int nrow = wn * 32 + lmod;
      v16h br0 = ldfrag(&cBr[nrow][0], k4, lhalf);
      v16h bi0 = ldfrag(&cBi[nrow][0], k4, lhalf);
      v16h bs0 = br0 + bi0;
      v16h br1 = ldfrag(&cBr[nrow + 16][0], k4, lhalf);
      v16h bi1 = ldfrag(&cBi[nrow + 16][0], k4, lhalf);
      v16h bs1 = br1 + bi1;
      p1a = WMMA_F16(ar, br0, p1a);
      p2a = WMMA_F16(ai, bi0, p2a);
      p3a = WMMA_F16(as, bs0, p3a);
      p1b = WMMA_F16(ar, br1, p1b);
      p2b = WMMA_F16(ai, bi1, p2b);
      p3b = WMMA_F16(as, bs1, p3b);
    }
    // ---- store staged regs into the alternate buffer ----
    if (hasNext) {
      int q = p ^ 1;
#pragma unroll
      for (int j = 0; j < 8; ++j) {
        sAr[q][srow][skg + 2 * j]     = (_Float16)ra[j].x;
        sAi[q][srow][skg + 2 * j]     = (_Float16)ra[j].y;
        sAr[q][srow][skg + 2 * j + 1] = (_Float16)ra[j].z;
        sAi[q][srow][skg + 2 * j + 1] = (_Float16)ra[j].w;
        sBr[q][srow][skg + 2 * j]     = (_Float16)rw[j].x;
        sBi[q][srow][skg + 2 * j]     = (_Float16)rw[j].y;
        sBr[q][srow][skg + 2 * j + 1] = (_Float16)rw[j].z;
        sBi[q][srow][skg + 2 * j + 1] = (_Float16)rw[j].w;
      }
    }
    __syncthreads();
    p ^= 1;
  }

  // ---- epilogue: Cr = p1 - p2, Ci = p3 - p1 - p2 ----
  v8f cr0 = p1a - p2a, ci0 = p3a - p1a - p2a;
  v8f cr1 = p1b - p2b, ci1 = p3b - p1b - p2b;
  int nbase = n0 + wn * 32 + lmod;
#pragma unroll
  for (int r = 0; r < 8; ++r) {
    int m = m0 + wm * 16 + lhalf * 8 + r;   // C layout: VGPR r -> M=r / M=8+r
    size_t c0 = (size_t)m * ldc + nbase;
    float2 o0 = make_float2(cr0[r], ci0[r]);
    float2 o1 = make_float2(cr1[r], ci1[r]);
    if (flags & 1) {
      float2 r0 = resid[c0], r1 = resid[c0 + 16];
      o0.x += r0.x; o0.y += r0.y;
      o1.x += r1.x; o1.y += r1.y;
    }
    if (flags & 2) {
      float2 b0 = bias[nbase], b1 = bias[nbase + 16];
      o0.x += b0.x; o0.y += b0.y;
      o1.x += b1.x; o1.y += b1.y;
    }
    C[c0] = o0;
    C[c0 + 16] = o1;
  }
}

// ---------------------------------------------------------------------------
// Attention logits: L[bh,q,k] = scale * Re( Q . conj(K) ) = scale*(qr.kr+qi.ki)
// Grid: (ktile=2, qtile=2, B*H). One K-chunk (DH=64). Already minimal: 2 WMMA
// per tile per k-step (real output only).
// ---------------------------------------------------------------------------
__global__ __launch_bounds__(256) void k_qk(const float2* __restrict__ Q,
                                            const float2* __restrict__ K,
                                            float* __restrict__ L) {
  __shared__ __attribute__((aligned(16))) _Float16 sQr[64][LKP], sQi[64][LKP];
  __shared__ __attribute__((aligned(16))) _Float16 sKr[64][LKP], sKi[64][LKP];

  int kt = blockIdx.x, qt = blockIdx.y, bh = blockIdx.z;
  int b = bh >> 4, hd = bh & 15;
  int tid = threadIdx.x, lane = tid & 31, wave = tid >> 5;
  int wm = wave & 3, wn = wave >> 2;
  int lmod = lane & 15, lhalf = lane >> 4;
  int srow = tid >> 2, skg = (tid & 3) * 16;

  const float2* gq = Q + (size_t)(b * SS + qt * 64 + srow) * DD + hd * DHD + skg;
  const float2* gk = K + (size_t)(b * SS + kt * 64 + srow) * DD + hd * DHD + skg;
#pragma unroll
  for (int j = 0; j < 16; ++j) {
    float2 vq = gq[j];
    sQr[srow][skg + j] = (_Float16)vq.x;
    sQi[srow][skg + j] = (_Float16)vq.y;
    float2 vk = gk[j];
    sKr[srow][skg + j] = (_Float16)vk.x;
    sKi[srow][skg + j] = (_Float16)vk.y;
  }
  __syncthreads();

  v8f c0 = {0, 0, 0, 0, 0, 0, 0, 0}, c1 = {0, 0, 0, 0, 0, 0, 0, 0};
#pragma unroll
  for (int ks = 0; ks < 2; ++ks) {
    int k4 = ks * 32;
    int arow = wm * 16 + lmod;
    v16h qr = ldfrag(&sQr[arow][0], k4, lhalf);
    v16h qi = ldfrag(&sQi[arow][0], k4, lhalf);
    int nrow = wn * 32 + lmod;
    v16h kr0 = ldfrag(&sKr[nrow][0], k4, lhalf);
    v16h ki0 = ldfrag(&sKi[nrow][0], k4, lhalf);
    v16h kr1 = ldfrag(&sKr[nrow + 16][0], k4, lhalf);
    v16h ki1 = ldfrag(&sKi[nrow + 16][0], k4, lhalf);
    c0 = WMMA_F16(qr, kr0, c0);
    c0 = WMMA_F16(qi, ki0, c0);
    c1 = WMMA_F16(qr, kr1, c1);
    c1 = WMMA_F16(qi, ki1, c1);
  }

  size_t lb = (size_t)bh * (SS * SS);
#pragma unroll
  for (int r = 0; r < 8; ++r) {
    int qm = qt * 64 + wm * 16 + lhalf * 8 + r;
    int kn = kt * 64 + wn * 32 + lmod;
    L[lb + (size_t)qm * SS + kn] = c0[r] * 0.125f;        // 1/sqrt(64)
    L[lb + (size_t)qm * SS + kn + 16] = c1[r] * 0.125f;
  }
}

// ---------------------------------------------------------------------------
// Row softmax over 128 logits. One block (128 thr) per (b,h,q) row. In place.
// ---------------------------------------------------------------------------
__global__ __launch_bounds__(128) void k_softmax(float* __restrict__ L) {
  float* p = L + (size_t)blockIdx.x * SS;
  int tid = threadIdx.x;
  float v = p[tid];
  __shared__ float sred[4];
  float m = v;
#pragma unroll
  for (int o = 16; o > 0; o >>= 1) m = fmaxf(m, __shfl_xor(m, o, 32));
  if ((tid & 31) == 0) sred[tid >> 5] = m;
  __syncthreads();
  m = fmaxf(fmaxf(sred[0], sred[1]), fmaxf(sred[2], sred[3]));
  __syncthreads();
  float e = __expf(v - m);
  float s = e;
#pragma unroll
  for (int o = 16; o > 0; o >>= 1) s += __shfl_xor(s, o, 32);
  if ((tid & 31) == 0) sred[tid >> 5] = s;
  __syncthreads();
  s = sred[0] + sred[1] + sred[2] + sred[3];
  p[tid] = e / s;
}

// ---------------------------------------------------------------------------
// O = A * V  (A real f32 SxS probs, V complex). Grid: (qtile=2, B*H).
// Output tile 64(q) x 64(dh), K=128 over key tokens (2 chunks). 2 WMMA/k-step
// per tile (real A), already minimal.
// ---------------------------------------------------------------------------
__global__ __launch_bounds__(256) void k_av(const float* __restrict__ A,
                                            const float2* __restrict__ V,
                                            float2* __restrict__ O) {
  __shared__ __attribute__((aligned(16))) _Float16 sA[64][LKP];
  __shared__ __attribute__((aligned(16))) _Float16 sVr[64][LKP], sVi[64][LKP];

  int qt = blockIdx.x, bh = blockIdx.y;
  int b = bh >> 4, hd = bh & 15;
  int tid = threadIdx.x, lane = tid & 31, wave = tid >> 5;
  int wm = wave & 3, wn = wave >> 2;
  int lmod = lane & 15, lhalf = lane >> 4;
  int srow = tid >> 2, skg = (tid & 3) * 16;

  v8f cr0 = {0, 0, 0, 0, 0, 0, 0, 0}, ci0 = {0, 0, 0, 0, 0, 0, 0, 0};
  v8f cr1 = {0, 0, 0, 0, 0, 0, 0, 0}, ci1 = {0, 0, 0, 0, 0, 0, 0, 0};

  for (int kc = 0; kc < SS; kc += LK) {
    const float* ga = A + (size_t)bh * (SS * SS) + (size_t)(qt * 64 + srow) * SS + kc + skg;
    const float2* gv = V + (size_t)(b * SS + kc + srow) * DD + hd * DHD + skg;
#pragma unroll
    for (int j = 0; j < 16; ++j) {
      sA[srow][skg + j] = (_Float16)ga[j];
      float2 vv = gv[j];
      sVr[skg + j][srow] = (_Float16)vv.x;   // transpose into [n=dh][k=key] layout
      sVi[skg + j][srow] = (_Float16)vv.y;
    }
    __syncthreads();
#pragma unroll
    for (int ks = 0; ks < 2; ++ks) {
      int k4 = ks * 32;
      int arow = wm * 16 + lmod;
      v16h a = ldfrag(&sA[arow][0], k4, lhalf);
      int nrow = wn * 32 + lmod;
      v16h br0 = ldfrag(&sVr[nrow][0], k4, lhalf);
      v16h bi0 = ldfrag(&sVi[nrow][0], k4, lhalf);
      v16h br1 = ldfrag(&sVr[nrow + 16][0], k4, lhalf);
      v16h bi1 = ldfrag(&sVi[nrow + 16][0], k4, lhalf);
      cr0 = WMMA_F16(a, br0, cr0);
      ci0 = WMMA_F16(a, bi0, ci0);
      cr1 = WMMA_F16(a, br1, cr1);
      ci1 = WMMA_F16(a, bi1, ci1);
    }
    __syncthreads();
  }

#pragma unroll
  for (int r = 0; r < 8; ++r) {
    int qm = qt * 64 + wm * 16 + lhalf * 8 + r;
    int n = wn * 32 + lmod;
    size_t ob = (size_t)(b * SS + qm) * DD + hd * DHD + n;
    O[ob] = make_float2(cr0[r], ci0[r]);
    O[ob + 16] = make_float2(cr1[r], ci1[r]);
  }
}

// ---------------------------------------------------------------------------
// Host-side orchestration.
// ---------------------------------------------------------------------------
extern "C" void kernel_launch(void* const* d_in, const int* in_sizes, int n_in,
                              void* d_out, int out_size, void* d_ws, size_t ws_size,
                              hipStream_t stream) {
  (void)in_sizes; (void)n_in; (void)out_size; (void)ws_size;

  const int*    x     = (const int*)d_in[0];
  const float2* emb   = (const float2*)d_in[1];
  const float2* enc_w = (const float2*)d_in[2];
  const float*  enc_g = (const float*)d_in[3];
  const float2* dec_w = (const float2*)d_in[4];
  const float2* dec_b = (const float2*)d_in[5];
  const float*  bn1   = (const float*)d_in[6];
  const float*  bn2   = (const float*)d_in[7];
  const float2* wq    = (const float2*)d_in[8];
  const float2* wk    = (const float2*)d_in[9];
  const float2* wv    = (const float2*)d_in[10];
  const float2* wo    = (const float2*)d_in[11];
  const float2* w1    = (const float2*)d_in[12];
  const float2* w2    = (const float2*)d_in[13];

  char* ws = (char*)d_ws;
  const size_t MB = 1ull << 20;
  float2* z  = (float2*)(ws + 0 * MB);     // (4096,1024) c64 = 32MB
  float2* h  = (float2*)(ws + 32 * MB);    // 32MB
  float2* qb = (float2*)(ws + 64 * MB);    // 32MB (also reused for attn output)
  float2* kb = (float2*)(ws + 96 * MB);    // 32MB
  float2* vb = (float2*)(ws + 128 * MB);   // 32MB
  float2* ff = (float2*)(ws + 160 * MB);   // (4096,4096) c64 = 128MB
  float*  lg = (float*)(ws + 288 * MB);    // (B,H,S,S) f32 = 32MB

  dim3 blk(256);
  dim3 gD(DD / 64, MTOK / 64);    // N=1024 GEMMs
  dim3 gF(FFD / 64, MTOK / 64);   // N=4096 GEMM

  // 1) embedding gather + rotary
  k_embed_rot<<<MTOK, blk, 0, stream>>>(x, emb, z);

  // 2) encoder: z = crms(z @ enc_w^T, enc_g)
  k_cgemm<<<gD, blk, 0, stream>>>(z, enc_w, nullptr, nullptr, h, DD, DD, DD, DD, 0);
  k_crms<<<MTOK, blk, 0, stream>>>(h, enc_g, z);

  for (int i = 0; i < NBK; ++i) {
    const size_t wD = (size_t)i * DD * DD;
    const size_t wF = (size_t)i * FFD * DD;
    // attention
    k_crms<<<MTOK, blk, 0, stream>>>(z, bn1 + i * DD, h);
    k_cgemm<<<gD, blk, 0, stream>>>(h, wq + wD, nullptr, nullptr, qb, DD, DD, DD, DD, 0);
    k_cgemm<<<gD, blk, 0, stream>>>(h, wk + wD, nullptr, nullptr, kb, DD, DD, DD, DD, 0);
    k_cgemm<<<gD, blk, 0, stream>>>(h, wv + wD, nullptr, nullptr, vb, DD, DD, DD, DD, 0);
    k_qk<<<dim3(2, 2, BB * HH), blk, 0, stream>>>(qb, kb, lg);
    k_softmax<<<BB * HH * SS, 128, 0, stream>>>(lg);
    k_av<<<dim3(2, BB * HH), blk, 0, stream>>>(lg, vb, qb);   // attn out -> qb
    k_cgemm<<<gD, blk, 0, stream>>>(qb, wo + wD, z, nullptr, z, DD, DD, DD, DD, 1);
    // feed-forward with phase twist
    k_crms<<<MTOK, blk, 0, stream>>>(z, bn2 + i * DD, h);
    k_cgemm<<<gF, blk, 0, stream>>>(h, w1 + wF, nullptr, nullptr, ff, DD, DD, DD, FFD, 0);
    k_phase<<<(MTOK * FFD) / 256, blk, 0, stream>>>(ff, (size_t)MTOK * FFD);
    k_cgemm<<<gD, blk, 0, stream>>>(ff, w2 + wF, z, nullptr, z, FFD, FFD, FFD, DD, 1);
  }

  // decoder: out = z @ dec_w^T + dec_b
  k_cgemm<<<gD, blk, 0, stream>>>(z, dec_w, nullptr, dec_b, (float2*)d_out,
                                  DD, DD, DD, DD, 2);
}
